// BicausalNet_4681514353009
// MI455X (gfx1250) — compile-verified, matrix operand order checked by reading, not run on hardware
//
#include <hip/hip_runtime.h>

#define EMB 384
#define NL 12
#define BB 4
#define LL 4096
#define NPOS (2*LL+1)      // 8193 positions in state tensor
#define NINT (2*LL-1)      // 8191 interior outputs per layer
#define TROWS 32                       // positions per workgroup
#define TILES ((NINT + TROWS - 1) / TROWS)   // 256 row tiles

typedef __attribute__((ext_vector_type(16))) __bf16 v16bf;
typedef __attribute__((ext_vector_type(8)))  float  v8f;

union FragA { v16bf v; uint4 q[2]; };

__device__ __forceinline__ unsigned short f2bf(float f) {
    unsigned u = __float_as_uint(f);
    unsigned r = 0x7FFFu + ((u >> 16) & 1u);      // round to nearest even
    return (unsigned short)((u + r) >> 16);
}

// ---------------------------------------------------------------- weights fp32 -> bf16
__global__ __launch_bounds__(256) void cvt_kernel(const float* __restrict__ w,
                                                  unsigned short* __restrict__ o, int n) {
    int i = blockIdx.x * 256 + threadIdx.x;
    if (i < n) o[i] = f2bf(w[i]);
}

// ---------------------------------------------------------------- build initial state
__global__ __launch_bounds__(256) void init_kernel(const float* __restrict__ embs,
                                                   const float* __restrict__ mask,
                                                   float* __restrict__ x) {
    size_t i = (size_t)blockIdx.x * 256 + threadIdx.x;
    const size_t total = (size_t)BB * NPOS * 2 * EMB;
    if (i >= total) return;
    int c = (int)(i % EMB); size_t r = i / EMB;
    int p = (int)(r % 2);   r /= 2;
    int pos = (int)(r % NPOS);
    int b   = (int)(r / NPOS);
    float val = mask[b * EMB + c];
    if (p == 0) {
        if (pos >= 1 && pos <= LL)   val = embs[((size_t)b * LL + (pos - 1)) * EMB + c];
        else if (pos == NPOS - 1)    val = embs[(size_t)b * LL * EMB + c];
    }
    x[i] = val;
}

// ---------------------------------------------------------------- fused layer kernel
// One WG = 32-position tile (two 16-row sub-tiles sharing B fragments).
// 12 waves x 2 N-tiles cover all 384 output channels.
// out[l,p] = relu(Wc x[l,p] + Wr x[l+1,0] + Wl x[l-1,0] + b) + x[l,p], stored scrambled.
#define RSd 196         // LDS row stride in dwords (= 392 bf16, conflict-free)
#define ROWS0 34        // p=0 rows: positions l0-1 .. l0+32
#define ROWS1 32        // p=1 rows: positions l0   .. l0+31
#define NTHREADS 384

__global__ __launch_bounds__(NTHREADS)
void layer_kernel(const float* __restrict__ xin, float* __restrict__ xout,
                  const unsigned short* __restrict__ wl,
                  const unsigned short* __restrict__ wc,
                  const unsigned short* __restrict__ wr,
                  const float* __restrict__ bias)
{
    __shared__ unsigned int lds[(ROWS0 + ROWS1) * RSd];   // 51,744 B
    const int t  = blockIdx.x;
    const int b  = blockIdx.y;
    const int l0 = 1 + t * TROWS;
    const int tid = threadIdx.x;

    // Stage A tiles to LDS as packed bf16 pairs: 66*192/384 = 33 iters, fully uniform.
    for (int idx = tid; idx < (ROWS0 + ROWS1) * (EMB / 2); idx += NTHREADS) {
        int row = idx / (EMB / 2), pr = idx % (EMB / 2);
        int pos, p;
        if (row < ROWS0) { pos = l0 - 1 + row;       p = 0; }
        else             { pos = l0 + (row - ROWS0); p = 1; }
        if (pos > NPOS - 1) pos = NPOS - 1;           // clamp for last tile
        const float2 f = *(const float2*)(xin + (((size_t)b * NPOS + pos) * 2 + p) * EMB + pr * 2);
        lds[row * RSd + pr] = (unsigned)f2bf(f.x) | ((unsigned)f2bf(f.y) << 16);
    }
    __syncthreads();

    const int lane = tid & 31;
    const int wave = tid >> 5;         // 0..11
    const int hi   = lane >> 4;        // K-half select for A/B fragments
    const int rl   = lane & 15;        // A row / B,D column within tile

    v8f accn[2][2], acc0[2][2], acc1[2][2];   // [n-tile][sub-tile]
    #pragma unroll
    for (int i = 0; i < 2; ++i)
        #pragma unroll
        for (int s = 0; s < 2; ++s) { accn[i][s] = {}; acc0[i][s] = {}; acc1[i][s] = {}; }

    for (int ks = 0; ks < EMB / 32; ++ks) {       // 12 K-steps of 32
        const int c0  = ks * 32 + hi * 8;         // bf16 column of first chunk
        const int dw0 = c0 >> 1;                  // dword offset within LDS row

        FragA a_l[2], a_c0[2], a_r[2], a_c1[2];
        #pragma unroll
        for (int s = 0; s < 2; ++s) {
            const int rb = rl + s * 16;
            const unsigned int* r0 = &lds[(rb + 0) * RSd + dw0];          // x[l-1,0]
            a_l[s].q[0]  = *(const uint4*)r0;     a_l[s].q[1]  = *(const uint4*)(r0 + 8);
            const unsigned int* r1 = &lds[(rb + 1) * RSd + dw0];          // x[l,0]
            a_c0[s].q[0] = *(const uint4*)r1;     a_c0[s].q[1] = *(const uint4*)(r1 + 8);
            const unsigned int* r2 = &lds[(rb + 2) * RSd + dw0];          // x[l+1,0]
            a_r[s].q[0]  = *(const uint4*)r2;     a_r[s].q[1]  = *(const uint4*)(r2 + 8);
            const unsigned int* r3 = &lds[(ROWS0 + rb) * RSd + dw0];      // x[l,1]
            a_c1[s].q[0] = *(const uint4*)r3;     a_c1[s].q[1] = *(const uint4*)(r3 + 8);
        }

        #pragma unroll
        for (int i = 0; i < 2; ++i) {
            const int d = (wave * 2 + i) * 16 + rl;       // output channel (B column)
            const size_t wo = (size_t)d * EMB + c0;
            FragA bl, bc, br;
            bl.q[0] = *(const uint4*)(wl + wo); bl.q[1] = *(const uint4*)(wl + wo + 16);
            bc.q[0] = *(const uint4*)(wc + wo); bc.q[1] = *(const uint4*)(wc + wo + 16);
            br.q[0] = *(const uint4*)(wr + wo); br.q[1] = *(const uint4*)(wr + wo + 16);

            #pragma unroll
            for (int s = 0; s < 2; ++s) {
                accn[i][s] = __builtin_amdgcn_wmma_f32_16x16x32_bf16(false, a_l[s].v,  false, bl.v, (short)0, accn[i][s], false, false);
                accn[i][s] = __builtin_amdgcn_wmma_f32_16x16x32_bf16(false, a_r[s].v,  false, br.v, (short)0, accn[i][s], false, false);
                acc0[i][s] = __builtin_amdgcn_wmma_f32_16x16x32_bf16(false, a_c0[s].v, false, bc.v, (short)0, acc0[i][s], false, false);
                acc1[i][s] = __builtin_amdgcn_wmma_f32_16x16x32_bf16(false, a_c1[s].v, false, bc.v, (short)0, acc1[i][s], false, false);
            }
        }
    }

    // Epilogue: bias + relu + residual, store to scrambled position.
    #pragma unroll
    for (int i = 0; i < 2; ++i) {
        const int d = (wave * 2 + i) * 16 + rl;
        const float bv = bias[d];
        #pragma unroll
        for (int s = 0; s < 2; ++s) {
            #pragma unroll
            for (int v = 0; v < 8; ++v) {
                const int m = v + 8 * hi;             // D layout: row = v + 8*(lane/16)
                const int l = l0 + s * 16 + m;
                if (l > NINT) continue;               // mask ragged last tile
                const int j = l - 1;
                const int k = (j & 1) ? (4097 + ((j - 1) >> 1)) : ((j >> 1) + 1);
                const size_t in0 = (((size_t)b * NPOS + l) * 2 + 0) * EMB + d;
                const float o0 = fmaxf(acc0[i][s][v] + accn[i][s][v] + bv, 0.f) + xin[in0];
                const float o1 = fmaxf(acc1[i][s][v] + accn[i][s][v] + bv, 0.f) + xin[in0 + EMB];
                const size_t oo = (((size_t)b * NPOS + k) * 2) * EMB + d;
                xout[oo] = o0;  xout[oo + EMB] = o1;
                if (j == 0) {                          // wrap pad: core[0] -> position 8192
                    const size_t oe = (((size_t)b * NPOS + (NPOS - 1)) * 2) * EMB + d;
                    xout[oe] = o0;  xout[oe + EMB] = o1;
                }
                if (j == NINT - 2) {                   // wrap pad: core[-1] (j=8189) -> position 0
                    const size_t oe = ((size_t)b * NPOS * 2) * EMB + d;
                    xout[oe] = o0;  xout[oe + EMB] = o1;
                }
            }
        }
    }
}

// ---------------------------------------------------------------- final extraction
// out0[b,i,c] = xF[b, 2i+1, 0, c] ; out1[b,i,c] = xF[b, 2i+1, 1, c]
__global__ __launch_bounds__(256) void extract_kernel(const float* __restrict__ x,
                                                      float* __restrict__ out) {
    size_t i = (size_t)blockIdx.x * 256 + threadIdx.x;
    const size_t total = (size_t)BB * LL * EMB;
    if (i >= total) return;
    int c = (int)(i % EMB); size_t r = i / EMB;
    int pos = (int)(r % LL);
    int b   = (int)(r / LL);
    const size_t src = (((size_t)b * NPOS + (2 * pos + 1)) * 2) * EMB + c;
    out[i]         = x[src];          // p = 0 half
    out[total + i] = x[src + EMB];    // p = 1 half
}

extern "C" void kernel_launch(void* const* d_in, const int* in_sizes, int n_in,
                              void* d_out, int out_size, void* d_ws, size_t ws_size,
                              hipStream_t stream) {
    const float* embs     = (const float*)d_in[0];
    const float* mask     = (const float*)d_in[1];
    const float* w_left   = (const float*)d_in[2];
    const float* w_center = (const float*)d_in[3];
    const float* w_right  = (const float*)d_in[4];
    const float* bias     = (const float*)d_in[5];
    float* out = (float*)d_out;

    // Workspace layout: two fp32 ping-pong state buffers + bf16 weight copies (~212 MB).
    char* ws = (char*)d_ws;
    const size_t XN = (size_t)BB * NPOS * 2 * EMB;            // elements per state buffer
    const size_t XB = XN * sizeof(float);
    float* x0 = (float*)ws;
    float* x1 = (float*)(ws + XB);
    unsigned short* wlb = (unsigned short*)(ws + 2 * XB);
    unsigned short* wcb = wlb + (size_t)NL * EMB * EMB;
    unsigned short* wrb = wcb + (size_t)NL * EMB * EMB;

    const int wn = NL * EMB * EMB;
    cvt_kernel<<<(wn + 255) / 256, 256, 0, stream>>>(w_left,   wlb, wn);
    cvt_kernel<<<(wn + 255) / 256, 256, 0, stream>>>(w_center, wcb, wn);
    cvt_kernel<<<(wn + 255) / 256, 256, 0, stream>>>(w_right,  wrb, wn);

    init_kernel<<<(unsigned)((XN + 255) / 256), 256, 0, stream>>>(embs, mask, x0);

    float* xa = x0;
    float* xb = x1;
    for (int layer = 0; layer < NL; ++layer) {
        dim3 grid(TILES, BB);
        layer_kernel<<<grid, NTHREADS, 0, stream>>>(xa, xb,
            wlb + (size_t)layer * EMB * EMB,
            wcb + (size_t)layer * EMB * EMB,
            wrb + (size_t)layer * EMB * EMB,
            bias + (size_t)layer * EMB);
        float* tmp = xa; xa = xb; xb = tmp;
    }

    const size_t on = (size_t)BB * LL * EMB;
    extract_kernel<<<(unsigned)((on + 255) / 256), 256, 0, stream>>>(xa, out);
}
